// VisionTransformer_88244398064141
// MI455X (gfx1250) — matrix operand order repaired and checked by reading, not compiled, not run on hardware
//
#include <hip/hip_runtime.h>
#include <hip/hip_bf16.h>
#include <cmath>

// ---------------- types ----------------
typedef __attribute__((ext_vector_type(16))) __bf16 v16bf;
typedef __attribute__((ext_vector_type(8)))  __bf16 v8bf;
typedef __attribute__((ext_vector_type(8)))  float  v8f;

// ---------------- model dims ----------------
static constexpr int BATCH  = 32;
static constexpr int SEQ    = 197;
static constexpr int DIM    = 768;
static constexpr int NH     = 12;
static constexpr int DHD    = 64;
static constexpr int NPAD   = 224;            // seq padded to multiple of 32 (WMMA K)
static constexpr int BHN    = BATCH * NH;     // 384
static constexpr int MLPD   = 3072;
static constexpr int NCLS   = 1000;
static constexpr int NPATCH = 196;
static constexpr int MTOK   = BATCH * SEQ;    // 6304
static constexpr int MPAT   = BATCH * NPATCH; // 6272
static constexpr int DEPTH  = 12;

// ---------------- helpers ----------------
__device__ __forceinline__ __bf16 f2bf(float f) {
  unsigned u = __float_as_uint(f);
  u += 0x7fffu + ((u >> 16) & 1u);               // round-to-nearest-even
  unsigned short s = (unsigned short)(u >> 16);
  return __builtin_bit_cast(__bf16, s);
}
__device__ __forceinline__ v8f vzero8() {
  v8f z;
#pragma unroll
  for (int i = 0; i < 8; ++i) z[i] = 0.0f;
  return z;
}
// build a v16bf fragment from two contiguous 8-element (16B) chunks
__device__ __forceinline__ v16bf pack16(const __bf16* lo, const __bf16* hi) {
  v8bf a = *(const v8bf*)lo;
  v8bf b = *(const v8bf*)hi;
  v16bf r;
#pragma unroll
  for (int i = 0; i < 8; ++i) { r[i] = a[i]; r[i + 8] = b[i]; }
  return r;
}
__device__ __forceinline__ v8f wmma_bf16(v16bf a, v16bf b, v8f c) {
  return __builtin_amdgcn_wmma_f32_16x16x32_bf16(false, a, false, b, (short)0, c,
                                                 false, false);
}

// CDNA5 async global->LDS copy (16B per lane), tracked by ASYNCcnt.
__device__ __forceinline__ void async_ld_b128(const __bf16* gsrc, __bf16* ldst) {
  unsigned l = (unsigned)(uintptr_t)ldst;   // low 32 bits = LDS byte offset
  asm volatile("global_load_async_to_lds_b128 %0, %1, off"
               :: "v"(l), "v"(gsrc) : "memory");
}
__device__ __forceinline__ void wait_async4() {
  asm volatile("s_wait_asynccnt 0x4" ::: "memory");
}
__device__ __forceinline__ void wait_async0() {
  asm volatile("s_wait_asynccnt 0x0" ::: "memory");
}

// =====================================================================
// Generic bf16 GEMM: C[M,Nc] = A[M,K] * BT[Nc,K]^T (+bias)(+res)(+GELU)
//   A row-major [M,K];  BT row-major [Nc,K] (weights pre-transposed).
// block = 256 threads (8 waves), tile 128(M) x 128(N), K step 32.
// waves: 4 along M x 2 along N; wave tile 32x64 -> 8 WMMA / K-step.
// Double-buffered LDS stages filled with GLOBAL_LOAD_ASYNC_TO_LDS_B128;
// out-of-range tile rows are address-clamped (their C rows/cols are never
// stored), keeping a constant 4 async ops per wave per stage.
// MODE: 0 f32+bias | 1 f32+bias+residual | 2 bf16+bias | 3 bf16+bias+GELU
// =====================================================================
template <int MODE>
__global__ __launch_bounds__(256)
void gemm_bf16_kernel(const __bf16* __restrict__ A, int lda,
                      const __bf16* __restrict__ BT, int ldb,
                      const float* __restrict__ bias,
                      const float* __restrict__ residual,
                      void* __restrict__ Out, int ldo,
                      int M, int Nc, int K)
{
  __shared__ __bf16 As[2][128][40];   // [stage][m][k] (+8 pad)
  __shared__ __bf16 Bs[2][128][40];   // [stage][n][k] (+8 pad)

  const int tid   = threadIdx.x;
  const int lane  = tid & 31;
  const int wave  = tid >> 5;
  const int waveM = wave & 3;      // 4 waves along M (32 rows each)
  const int waveN = wave >> 2;     // 2 waves along N (64 cols each)
  const int bM    = blockIdx.x * 128;
  const int bN    = blockIdx.y * 128;
  const int lidx  = lane & 15;
  const int lhal  = lane >> 4;

  v8f acc[2][4];
#pragma unroll
  for (int i = 0; i < 2; ++i)
#pragma unroll
    for (int j = 0; j < 4; ++j) acc[i][j] = vzero8();

  // per-stage copy assignment: 4 x 16B chunks per thread
  const int crow = tid >> 2;           // 0..63
  const int cko  = (tid & 3) * 8;      // 0,8,16,24
  const int ar0  = min(bM + crow,      M - 1);
  const int ar1  = min(bM + crow + 64, M - 1);
  const int br0  = min(bN + crow,      Nc - 1);
  const int br1  = min(bN + crow + 64, Nc - 1);

  auto issue = [&](int k0, int buf) {
    async_ld_b128(A  + (size_t)ar0 * lda + k0 + cko, &As[buf][crow][cko]);
    async_ld_b128(A  + (size_t)ar1 * lda + k0 + cko, &As[buf][crow + 64][cko]);
    async_ld_b128(BT + (size_t)br0 * ldb + k0 + cko, &Bs[buf][crow][cko]);
    async_ld_b128(BT + (size_t)br1 * ldb + k0 + cko, &Bs[buf][crow + 64][cko]);
  };

  const int nT = K >> 5;
  issue(0, 0);
  for (int t = 0; t < nT; ++t) {
    const int buf = t & 1;
    if (t + 1 < nT) { issue((t + 1) << 5, buf ^ 1); wait_async4(); }
    else            { wait_async0(); }
    __syncthreads();

    v16bf af[2], bfv[4];
#pragma unroll
    for (int i = 0; i < 2; ++i) {
      const int m = waveM * 32 + i * 16 + lidx;
      af[i] = pack16(&As[buf][m][lhal * 8], &As[buf][m][16 + lhal * 8]);
    }
#pragma unroll
    for (int j = 0; j < 4; ++j) {
      const int n = waveN * 64 + j * 16 + lidx;
      bfv[j] = pack16(&Bs[buf][n][lhal * 8], &Bs[buf][n][16 + lhal * 8]);
    }
#pragma unroll
    for (int i = 0; i < 2; ++i)
#pragma unroll
      for (int j = 0; j < 4; ++j)
        acc[i][j] = wmma_bf16(af[i], bfv[j], acc[i][j]);
    __syncthreads();
  }

  // ---- epilogue: D layout: lane<16 -> M=r, lane>=16 -> M=8+r; N = lane&15
#pragma unroll
  for (int i = 0; i < 2; ++i) {
#pragma unroll
    for (int j = 0; j < 4; ++j) {
      const int n  = bN + waveN * 64 + j * 16 + lidx;
      const int mb = bM + waveM * 32 + i * 16 + lhal * 8;
#pragma unroll
      for (int r = 0; r < 8; ++r) {
        const int m = mb + r;
        if (m < M && n < Nc) {
          float v = acc[i][j][r];
          if (bias) v += bias[n];
          if (MODE == 3) v = 0.5f * v * (1.0f + erff(v * 0.70710678118654752f));
          if (MODE == 1) v += residual[(size_t)m * ldo + n];
          if (MODE == 0 || MODE == 1)
            ((float*)Out)[(size_t)m * ldo + n] = v;
          else
            ((__bf16*)Out)[(size_t)m * ldo + n] = f2bf(v);
        }
      }
    }
  }
}

// =====================================================================
// Attention scores: S[bh,224,224] = scale * Q[bh,224,64] @ K[bh,224,64]^T
// one wave per 16x16 tile; fragments loaded straight from global (dh=64)
// =====================================================================
__global__ __launch_bounds__(32)
void attn_scores_kernel(const __bf16* __restrict__ Q, const __bf16* __restrict__ Kmat,
                        float* __restrict__ S, float scale)
{
  const int bh   = blockIdx.y;
  const int mt   = blockIdx.x / 14;
  const int nt   = blockIdx.x % 14;
  const int lane = threadIdx.x & 31;
  const int lidx = lane & 15, lhal = lane >> 4;

  const __bf16* qrow = Q    + ((size_t)bh * NPAD + mt * 16 + lidx) * DHD;
  const __bf16* krow = Kmat + ((size_t)bh * NPAD + nt * 16 + lidx) * DHD;

  v8f acc = vzero8();
#pragma unroll
  for (int c = 0; c < 2; ++c) {   // dh = 64 = 2 K-chunks of 32
    v16bf a = pack16(qrow + c * 32 + lhal * 8,  qrow + c * 32 + 16 + lhal * 8);
    v16bf b = pack16(krow + c * 32 + lhal * 8, krow + c * 32 + 16 + lhal * 8);
    acc = wmma_bf16(a, b, acc);
  }
  float* srow = S + ((size_t)bh * NPAD + mt * 16 + lhal * 8) * NPAD + nt * 16 + lidx;
#pragma unroll
  for (int r = 0; r < 8; ++r) srow[(size_t)r * NPAD] = acc[r] * scale;
}

// =====================================================================
// Row softmax over the real 197 columns; pad columns written as 0 (bf16)
// =====================================================================
__global__ __launch_bounds__(256)
void softmax_kernel(const float* __restrict__ S, __bf16* __restrict__ P)
{
  __shared__ float red[256];
  const int t = threadIdx.x;
  const float* srow = S + (size_t)blockIdx.x * NPAD;
  const float v = (t < SEQ) ? srow[t] : -3.0e38f;
  red[t] = v;
  __syncthreads();
  for (int s = 128; s > 0; s >>= 1) {
    if (t < s) red[t] = fmaxf(red[t], red[t + s]);
    __syncthreads();
  }
  const float mx = red[0];
  __syncthreads();
  const float e = (t < SEQ) ? expf(v - mx) : 0.0f;
  red[t] = e;
  __syncthreads();
  for (int s = 128; s > 0; s >>= 1) {
    if (t < s) red[t] += red[t + s];
    __syncthreads();
  }
  const float inv = 1.0f / red[0];
  if (t < NPAD)
    P[(size_t)blockIdx.x * NPAD + t] = f2bf((t < SEQ) ? e * inv : 0.0f);
}

// =====================================================================
// O[b,m,h*64+d] = P[bh,224,224] @ V; V pre-transposed: Vt[bh,64,224].
// One wave per 16x16 tile, both fragments contiguous from global.
// grid = (14 mtiles * 4 ntiles, BHN)
// =====================================================================
__global__ __launch_bounds__(32)
void attn_av_kernel(const __bf16* __restrict__ P, const __bf16* __restrict__ Vt,
                    __bf16* __restrict__ O)
{
  const int bh = blockIdx.y;
  const int b  = bh / NH, h = bh % NH;
  const int mt = blockIdx.x >> 2;   // 0..13
  const int nt = blockIdx.x & 3;    // 0..3
  const int lane = threadIdx.x & 31;
  const int lidx = lane & 15, lhal = lane >> 4;

  const __bf16* prow = P  + ((size_t)bh * NPAD + mt * 16 + lidx) * NPAD;
  const __bf16* vrow = Vt + ((size_t)bh * DHD  + nt * 16 + lidx) * NPAD;

  v8f acc = vzero8();
#pragma unroll
  for (int kc = 0; kc < 7; ++kc) {  // K = 224 = 7 chunks
    v16bf a  = pack16(prow + kc * 32 + lhal * 8,  prow + kc * 32 + 16 + lhal * 8);
    v16bf bb = pack16(vrow + kc * 32 + lhal * 8, vrow + kc * 32 + 16 + lhal * 8);
    acc = wmma_bf16(a, bb, acc);
  }
#pragma unroll
  for (int r = 0; r < 8; ++r) {
    const int m = mt * 16 + lhal * 8 + r;
    if (m < SEQ)
      O[((size_t)b * SEQ + m) * DIM + h * DHD + nt * 16 + lidx] = f2bf(acc[r]);
  }
}

// =====================================================================
// LayerNorm over 768 features, one block per row, bf16 output
// =====================================================================
__global__ __launch_bounds__(256)
void layernorm_kernel(const float* __restrict__ X, size_t rowStride,
                      const float* __restrict__ w, const float* __restrict__ b,
                      __bf16* __restrict__ Y, int ldy)
{
  __shared__ float red[256];
  const int t = threadIdx.x;
  const float* x = X + (size_t)blockIdx.x * rowStride;
  const float v0 = x[t], v1 = x[t + 256], v2 = x[t + 512];
  red[t] = v0 + v1 + v2;
  __syncthreads();
  for (int s = 128; s > 0; s >>= 1) { if (t < s) red[t] += red[t + s]; __syncthreads(); }
  const float mean = red[0] * (1.0f / 768.0f);
  __syncthreads();
  const float d0 = v0 - mean, d1 = v1 - mean, d2 = v2 - mean;
  red[t] = d0 * d0 + d1 * d1 + d2 * d2;
  __syncthreads();
  for (int s = 128; s > 0; s >>= 1) { if (t < s) red[t] += red[t + s]; __syncthreads(); }
  const float rstd = rsqrtf(red[0] * (1.0f / 768.0f) + 1e-6f);
  __bf16* y = Y + (size_t)blockIdx.x * ldy;
  y[t]       = f2bf(d0 * rstd * w[t]       + b[t]);
  y[t + 256] = f2bf(d1 * rstd * w[t + 256] + b[t + 256]);
  y[t + 512] = f2bf(d2 * rstd * w[t + 512] + b[t + 512]);
}

// ---------------- elementwise / data-movement kernels ----------------
// f32 [L][K][N] -> bf16 transposed [L][N][K]
__global__ void cvt_transpose_kernel(const float* __restrict__ src,
                                     __bf16* __restrict__ dst,
                                     int K, int N, long total)
{
  long i = (long)blockIdx.x * blockDim.x + threadIdx.x;
  if (i >= total) return;
  const long kn = (long)K * N;
  const long l  = i / kn;
  const int rem = (int)(i - l * kn);
  const int n = rem / K, k = rem - n * K;
  dst[i] = f2bf(src[(size_t)l * kn + (size_t)k * N + n]);
}

__global__ void patchify_kernel(const float* __restrict__ x, __bf16* __restrict__ Ap)
{
  long i = (long)blockIdx.x * blockDim.x + threadIdx.x;
  if (i >= (long)MPAT * DIM) return;
  const int col = (int)(i % DIM);
  const int p   = (int)((i / DIM) % NPATCH);
  const int b   = (int)(i / ((long)DIM * NPATCH));
  const int c = col >> 8, rem = col & 255, py = rem >> 4, px = rem & 15;
  const int gy = p / 14, gx = p % 14;
  const float v = x[(((size_t)b * 3 + c) * 224 + gy * 16 + py) * 224 + gx * 16 + px];
  Ap[i] = f2bf(v);
}

__global__ void assemble_h_kernel(const float* __restrict__ hp,
                                  const float* __restrict__ cls,
                                  const float* __restrict__ pos,
                                  float* __restrict__ h)
{
  long i = (long)blockIdx.x * blockDim.x + threadIdx.x;
  if (i >= (long)MTOK * DIM) return;
  const int d = (int)(i % DIM);
  const int n = (int)((i / DIM) % SEQ);
  const int b = (int)(i / ((long)DIM * SEQ));
  const float v = (n == 0) ? cls[d] : hp[((size_t)b * NPATCH + (n - 1)) * DIM + d];
  h[i] = v + pos[(size_t)n * DIM + d];
}

// qkv[B*SEQ, 2304] -> Q/K [bh,224,64] (pad rows 0) and Vt [bh,64,224]
__global__ void qkv_scatter_kernel(const __bf16* __restrict__ qkv,
                                   __bf16* __restrict__ Q, __bf16* __restrict__ Kd,
                                   __bf16* __restrict__ Vt)
{
  long i = (long)blockIdx.x * blockDim.x + threadIdx.x;
  if (i >= (long)BHN * NPAD * DHD) return;
  const int d  = (int)(i & 63);
  const int n  = (int)((i >> 6) % NPAD);
  const int bh = (int)(i / ((long)NPAD * DHD));
  const int b = bh / NH, h = bh % NH;
  __bf16 q = f2bf(0.0f), k = f2bf(0.0f), v = f2bf(0.0f);
  if (n < SEQ) {
    const size_t s = ((size_t)(b * SEQ + n)) * (3 * DIM) + h * DHD + d;
    q = qkv[s]; k = qkv[s + DIM]; v = qkv[s + 2 * DIM];
  }
  Q[i] = q; Kd[i] = k;
  Vt[((size_t)bh * DHD + d) * NPAD + n] = v;
}

// =====================================================================
// Host orchestration
// =====================================================================
extern "C" void kernel_launch(void* const* d_in, const int* in_sizes, int n_in,
                              void* d_out, int out_size, void* d_ws, size_t ws_size,
                              hipStream_t stream)
{
  (void)in_sizes; (void)n_in; (void)out_size; (void)ws_size;
  const float* x       = (const float*)d_in[0];
  const float* patch_w = (const float*)d_in[1];
  const float* patch_b = (const float*)d_in[2];
  const float* cls_tok = (const float*)d_in[3];
  const float* pos_emb = (const float*)d_in[4];
  const float* ln1_w   = (const float*)d_in[5];
  const float* ln1_b   = (const float*)d_in[6];
  const float* qkv_w   = (const float*)d_in[7];
  const float* qkv_b   = (const float*)d_in[8];
  const float* proj_w  = (const float*)d_in[9];
  const float* proj_b  = (const float*)d_in[10];
  const float* ln2_w   = (const float*)d_in[11];
  const float* ln2_b   = (const float*)d_in[12];
  const float* fc1_w   = (const float*)d_in[13];
  const float* fc1_b   = (const float*)d_in[14];
  const float* fc2_w   = (const float*)d_in[15];
  const float* fc2_b   = (const float*)d_in[16];
  const float* normf_w = (const float*)d_in[17];
  const float* normf_b = (const float*)d_in[18];
  const float* head_w  = (const float*)d_in[19];
  const float* head_b  = (const float*)d_in[20];

  size_t off = 0;
  auto alloc = [&](size_t bytes) -> void* {
    off = (off + 255) & ~(size_t)255;
    void* p = (char*)d_ws + off;
    off += bytes;
    return p;
  };

  // transposed bf16 weights: [N][K] per layer
  __bf16* wt_patch = (__bf16*)alloc((size_t)DIM * DIM * 2);
  __bf16* wt_qkv   = (__bf16*)alloc((size_t)DEPTH * DIM * 3 * DIM * 2);
  __bf16* wt_proj  = (__bf16*)alloc((size_t)DEPTH * DIM * DIM * 2);
  __bf16* wt_fc1   = (__bf16*)alloc((size_t)DEPTH * DIM * MLPD * 2);
  __bf16* wt_fc2   = (__bf16*)alloc((size_t)DEPTH * MLPD * DIM * 2);
  __bf16* wt_head  = (__bf16*)alloc((size_t)DIM * NCLS * 2);
  __bf16* Ap   = (__bf16*)alloc((size_t)MPAT * DIM * 2);
  float*  hp   = (float*) alloc((size_t)MPAT * DIM * 4);
  float*  h    = (float*) alloc((size_t)MTOK * DIM * 4);
  __bf16* y    = (__bf16*)alloc((size_t)MTOK * DIM * 2);
  __bf16* qkvb = (__bf16*)alloc((size_t)MTOK * 3 * DIM * 2);
  __bf16* Qb   = (__bf16*)alloc((size_t)BHN * NPAD * DHD * 2);
  __bf16* Kb   = (__bf16*)alloc((size_t)BHN * NPAD * DHD * 2);
  __bf16* Vt   = (__bf16*)alloc((size_t)BHN * DHD * NPAD * 2);
  float*  S    = (float*) alloc((size_t)BHN * NPAD * NPAD * 4);
  __bf16* Pm   = (__bf16*)alloc((size_t)BHN * NPAD * NPAD * 2);
  __bf16* O    = (__bf16*)alloc((size_t)MTOK * DIM * 2);
  __bf16* U    = (__bf16*)alloc((size_t)MTOK * MLPD * 2);
  __bf16* ycls = (__bf16*)alloc((size_t)BATCH * DIM * 2);

  auto cvtT = [&](const float* s, __bf16* d, int K, int N, long L) {
    const long tot = L * (long)K * N;
    cvt_transpose_kernel<<<(unsigned)((tot + 255) / 256), 256, 0, stream>>>(s, d, K, N, tot);
  };
  cvtT(patch_w, wt_patch, DIM, DIM, 1);
  cvtT(qkv_w,  wt_qkv,  DIM, 3 * DIM, DEPTH);
  cvtT(proj_w, wt_proj, DIM, DIM, DEPTH);
  cvtT(fc1_w,  wt_fc1,  DIM, MLPD, DEPTH);
  cvtT(fc2_w,  wt_fc2,  MLPD, DIM, DEPTH);
  cvtT(head_w, wt_head, DIM, NCLS, 1);

  patchify_kernel<<<(unsigned)(((long)MPAT * DIM + 255) / 256), 256, 0, stream>>>(x, Ap);

  auto gdim = [](int M, int Nc) { return dim3((M + 127) / 128, (Nc + 127) / 128); };

  // patch embed
  gemm_bf16_kernel<0><<<gdim(MPAT, DIM), 256, 0, stream>>>(
      Ap, DIM, wt_patch, DIM, patch_b, nullptr, hp, DIM, MPAT, DIM, DIM);
  assemble_h_kernel<<<(unsigned)(((long)MTOK * DIM + 255) / 256), 256, 0, stream>>>(
      hp, cls_tok, pos_emb, h);

  const float scale = 0.125f;  // (D/H)^-0.5 = 64^-0.5

  for (int l = 0; l < DEPTH; ++l) {
    layernorm_kernel<<<MTOK, 256, 0, stream>>>(h, (size_t)DIM, ln1_w + l * DIM,
                                               ln1_b + l * DIM, y, DIM);
    gemm_bf16_kernel<2><<<gdim(MTOK, 3 * DIM), 256, 0, stream>>>(
        y, DIM, wt_qkv + (size_t)l * DIM * 3 * DIM, DIM, qkv_b + l * 3 * DIM,
        nullptr, qkvb, 3 * DIM, MTOK, 3 * DIM, DIM);
    qkv_scatter_kernel<<<(unsigned)(((long)BHN * NPAD * DHD + 255) / 256), 256, 0,
                         stream>>>(qkvb, Qb, Kb, Vt);
    attn_scores_kernel<<<dim3(196, BHN), 32, 0, stream>>>(Qb, Kb, S, scale);
    softmax_kernel<<<BHN * NPAD, 256, 0, stream>>>(S, Pm);
    attn_av_kernel<<<dim3(56, BHN), 32, 0, stream>>>(Pm, Vt, O);
    gemm_bf16_kernel<1><<<gdim(MTOK, DIM), 256, 0, stream>>>(
        O, DIM, wt_proj + (size_t)l * DIM * DIM, DIM, proj_b + l * DIM, h, h, DIM,
        MTOK, DIM, DIM);
    layernorm_kernel<<<MTOK, 256, 0, stream>>>(h, (size_t)DIM, ln2_w + l * DIM,
                                               ln2_b + l * DIM, y, DIM);
    gemm_bf16_kernel<3><<<gdim(MTOK, MLPD), 256, 0, stream>>>(
        y, DIM, wt_fc1 + (size_t)l * DIM * MLPD, DIM, fc1_b + l * MLPD, nullptr,
        U, MLPD, MTOK, MLPD, DIM);
    gemm_bf16_kernel<1><<<gdim(MTOK, DIM), 256, 0, stream>>>(
        U, MLPD, wt_fc2 + (size_t)l * MLPD * DIM, MLPD, fc2_b + l * DIM, h, h, DIM,
        MTOK, DIM, MLPD);
  }

  layernorm_kernel<<<BATCH, 256, 0, stream>>>(h, (size_t)SEQ * DIM, normf_w, normf_b,
                                              ycls, DIM);
  gemm_bf16_kernel<0><<<gdim(BATCH, NCLS), 256, 0, stream>>>(
      ycls, DIM, wt_head, DIM, head_b, nullptr, (float*)d_out, NCLS,
      BATCH, NCLS, DIM);
}
